// EncoderLayer_88012469829904
// MI455X (gfx1250) — compile-verified, matrix operand order checked by reading, not run on hardware
//
#include <hip/hip_runtime.h>
#include <hip/hip_bf16.h>
#include <cstdint>

typedef __attribute__((ext_vector_type(16))) __bf16 v16bf;
typedef __attribute__((ext_vector_type(8)))  __bf16 v8bf;
typedef __attribute__((ext_vector_type(8)))  float  v8f;

#define B_   4
#define S_   1024
#define D_   1024
#define H_   16
#define DK_  64
#define FFN_ 4096

// ---------- helpers ----------

__device__ __forceinline__ __bf16 f2bf(float f) {
  union { float f; unsigned u; } v; v.f = f;
  unsigned r = v.u + 0x7FFFu + ((v.u >> 16) & 1u);
  unsigned short h = (unsigned short)(r >> 16);
  __bf16 o; __builtin_memcpy(&o, &h, 2); return o;
}

// Low 32 bits of the generic address of a __shared__ object == LDS byte offset
// (flat shared-aperture mapping uses addr[31:0] as the LDS address).
__device__ __forceinline__ unsigned lds_addr(const void* p) {
  return (unsigned)(uintptr_t)p;
}

// Async direct global->LDS copy, 16B per lane, tracked by ASYNCcnt.
// saddr form: global_load_async_to_lds_b128 vdst_lds, vaddr32, s[base]
__device__ __forceinline__ void async_copy16(unsigned ldsOff, unsigned gByteOff,
                                             const void* base) {
  asm volatile("global_load_async_to_lds_b128 %0, %1, %2"
               :: "v"(ldsOff), "v"(gByteOff), "s"(base) : "memory");
}

__device__ __forceinline__ void wait_async0() {
  asm volatile("s_wait_asynccnt 0" ::: "memory");
}

// Load a 16x32 bf16 WMMA fragment (A layout; also B when source is Bt[N][K])
// from a row-major [rows][ldk] bf16 tile in LDS.
__device__ __forceinline__ v16bf load_frag(const __bf16* base, int row0, int k0, int ldk) {
  const int lane = threadIdx.x & 31;
  const __bf16* p = base + (row0 + (lane & 15)) * ldk + k0 + ((lane >> 4) << 3);
  v8bf lo = *(const v8bf*)p;
  v8bf hi = *(const v8bf*)(p + 16);
  v16bf r;
#pragma unroll
  for (int i = 0; i < 8; ++i) { r[i] = lo[i]; r[i + 8] = hi[i]; }
  return r;
}

__device__ __forceinline__ v8f wmma_bf16(v16bf a, v16bf b, v8f c) {
  return __builtin_amdgcn_wmma_f32_16x16x32_bf16(false, a, false, b, (short)0, c,
                                                 false, false);
}

// ---------- LayerNorm (D=1024), outputs f32 and/or bf16 ----------

__global__ __launch_bounds__(256) void ln_kernel(
    const float* __restrict__ in, const float* __restrict__ g,
    const float* __restrict__ bt, float* __restrict__ outF,
    __bf16* __restrict__ outB)
{
  const int row = blockIdx.x;
  const int tid = threadIdx.x;
  const float* p = in + (size_t)row * D_;
  float v[4]; float s = 0.f, sq = 0.f;
#pragma unroll
  for (int i = 0; i < 4; ++i) {
    v[i] = p[tid + i * 256];
    s += v[i]; sq += v[i] * v[i];
  }
#pragma unroll
  for (int off = 16; off >= 1; off >>= 1) {
    s  += __shfl_xor(s,  off, 32);
    sq += __shfl_xor(sq, off, 32);
  }
  __shared__ float red[16];
  const int wave = tid >> 5, lane = tid & 31;
  if (lane == 0) { red[wave] = s; red[wave + 8] = sq; }
  __syncthreads();
  if (tid < 32) {
    float a = (tid < 8) ? red[tid] : 0.f;
    float c = (tid < 8) ? red[tid + 8] : 0.f;
#pragma unroll
    for (int off = 4; off >= 1; off >>= 1) {
      a += __shfl_xor(a, off, 32);
      c += __shfl_xor(c, off, 32);
    }
    if (tid == 0) { red[0] = a; red[8] = c; }
  }
  __syncthreads();
  const float mean = red[0] * (1.f / D_);
  const float var  = red[8] * (1.f / D_) - mean * mean;
  const float rstd = rsqrtf(var + 1e-5f);
#pragma unroll
  for (int i = 0; i < 4; ++i) {
    const int col = tid + i * 256;
    const float o = (v[i] - mean) * rstd * g[col] + bt[col];
    if (outF) outF[(size_t)row * D_ + col] = o;
    if (outB) outB[(size_t)row * D_ + col] = f2bf(o);
  }
}

// ---------- fp32 [K,N] -> bf16 transposed [N,K] ----------

__global__ __launch_bounds__(256) void transpose_f32_to_bf16(
    const float* __restrict__ W, __bf16* __restrict__ Wt, int K, int N)
{
  __shared__ float t[32][33];
  const int n0 = blockIdx.x * 32, k0 = blockIdx.y * 32;
  const int tx = threadIdx.x, ty = threadIdx.y;  // block (32, 8)
#pragma unroll
  for (int i = 0; i < 4; ++i)
    t[ty + i * 8][tx] = W[(size_t)(k0 + ty + i * 8) * N + (n0 + tx)];
  __syncthreads();
#pragma unroll
  for (int i = 0; i < 4; ++i)
    Wt[(size_t)(n0 + ty + i * 8) * K + (k0 + tx)] = f2bf(t[tx][ty + i * 8]);
}

// ---------- GEMM: C[M,N] = epi( A[M,K]bf16 @ Bt[N,K]bf16^T ) ----------
// 128x128 block tile, BK=32, 8 waves -> 32x64 per wave (2x4 WMMA frags).
// K-panels double-buffered in LDS via GLOBAL_LOAD_ASYNC_TO_LDS (ASYNCcnt).
// Main loop unrolled 2 K-steps so buffer indices are compile-time constants.
// Requires ksteps even (all K here are multiples of 64).

__global__ __launch_bounds__(256) void gemm_bf16_kernel(
    const __bf16* __restrict__ A, const __bf16* __restrict__ Bt,
    const float* __restrict__ bias, const float* __restrict__ res1,
    const float* __restrict__ res2, float* __restrict__ outF,
    __bf16* __restrict__ outB, int M, int N, int K, int doRelu)
{
  __shared__ __align__(16) __bf16 As[2][128 * 40];  // [row][k], +8 pad
  __shared__ __align__(16) __bf16 Bs[2][128 * 40];  // [n][k],  +8 pad
  const int tid  = threadIdx.x;
  const int lane = tid & 31, wave = tid >> 5;
  const int wm = wave >> 1, wn = wave & 1;
  const int m0 = wm * 32, n0 = wn * 64;
  const int rowM = blockIdx.y * 128, colN = blockIdx.x * 128;

  // per-thread 16B chunks of the 128x32 tiles (512 chunks, 2 per thread)
  const int c0 = tid,       r0 = c0 >> 2, q0 = c0 & 3;
  const int c1 = tid + 256, r1 = c1 >> 2, q1 = c1 & 3;
  const unsigned offA0 = (unsigned)(((rowM + r0) * K + q0 * 8) * 2);
  const unsigned offA1 = (unsigned)(((rowM + r1) * K + q1 * 8) * 2);
  const unsigned offB0 = (unsigned)(((colN + r0) * K + q0 * 8) * 2);
  const unsigned offB1 = (unsigned)(((colN + r1) * K + q1 * 8) * 2);
  unsigned ldsA[2][2], ldsB[2][2];
#pragma unroll
  for (int bufi = 0; bufi < 2; ++bufi) {
    ldsA[bufi][0] = lds_addr(&As[bufi][r0 * 40 + q0 * 8]);
    ldsA[bufi][1] = lds_addr(&As[bufi][r1 * 40 + q1 * 8]);
    ldsB[bufi][0] = lds_addr(&Bs[bufi][r0 * 40 + q0 * 8]);
    ldsB[bufi][1] = lds_addr(&Bs[bufi][r1 * 40 + q1 * 8]);
  }

  const v8f zero8 = {0.f, 0.f, 0.f, 0.f, 0.f, 0.f, 0.f, 0.f};
  v8f acc[2][4];
#pragma unroll
  for (int i = 0; i < 2; ++i)
#pragma unroll
    for (int j = 0; j < 4; ++j) acc[i][j] = zero8;

  const int ksteps = K >> 5;

#define ISSUE_STAGE(BUFI, KB)                                   \
  do {                                                          \
    const __bf16* aP = A  + ((size_t)(KB) << 5);                \
    const __bf16* bP = Bt + ((size_t)(KB) << 5);                \
    async_copy16(ldsA[BUFI][0], offA0, aP);                     \
    async_copy16(ldsA[BUFI][1], offA1, aP);                     \
    async_copy16(ldsB[BUFI][0], offB0, bP);                     \
    async_copy16(ldsB[BUFI][1], offB1, bP);                     \
  } while (0)

#define COMPUTE_PANEL(BUFI)                                     \
  do {                                                          \
    v16bf af[2], bfr[4];                                        \
    _Pragma("unroll")                                           \
    for (int i = 0; i < 2; ++i)                                 \
      af[i] = load_frag(As[BUFI], m0 + i * 16, 0, 40);          \
    _Pragma("unroll")                                           \
    for (int j = 0; j < 4; ++j)                                 \
      bfr[j] = load_frag(Bs[BUFI], n0 + j * 16, 0, 40);         \
    _Pragma("unroll")                                           \
    for (int i = 0; i < 2; ++i)                                 \
      _Pragma("unroll")                                         \
      for (int j = 0; j < 4; ++j)                               \
        acc[i][j] = wmma_bf16(af[i], bfr[j], acc[i][j]);        \
  } while (0)

  ISSUE_STAGE(0, 0);
  for (int kb = 0; kb < ksteps; kb += 2) {
    // ---- half step A: consume buffer 0, prefetch into buffer 1 ----
    wait_async0();                  // my async writes into buf0 are in LDS
    __syncthreads();                // all waves' writes visible; prior reads done
    if (kb + 1 < ksteps) ISSUE_STAGE(1, kb + 1);
    if (kb + 2 < ksteps) {
      __builtin_prefetch(A  + (size_t)(rowM + r0) * K + ((kb + 2) << 5), 0, 1);
      __builtin_prefetch(Bt + (size_t)(colN + r0) * K + ((kb + 2) << 5), 0, 1);
    }
    COMPUTE_PANEL(0);
    __syncthreads();                // reads of buf0 done before it is rewritten

    // ---- half step B: consume buffer 1, prefetch into buffer 0 ----
    wait_async0();
    __syncthreads();
    if (kb + 2 < ksteps) ISSUE_STAGE(0, kb + 2);
    COMPUTE_PANEL(1);
    __syncthreads();
  }
#undef ISSUE_STAGE
#undef COMPUTE_PANEL

  // epilogue: C/D layout lane=N (lane&15), vgpr r = M offset (+8 for lanes>=16)
  const int ho = (lane < 16) ? 0 : 8;
  const int nc = lane & 15;
#pragma unroll
  for (int i = 0; i < 2; ++i)
#pragma unroll
    for (int j = 0; j < 4; ++j)
#pragma unroll
      for (int r = 0; r < 8; ++r) {
        const size_t gr = (size_t)(rowM + m0 + i * 16 + r + ho);
        const size_t gc = (size_t)(colN + n0 + j * 16 + nc);
        const size_t idx = gr * (size_t)N + gc;
        float v = acc[i][j][r];
        if (bias) v += bias[gc];
        if (res1) v += res1[idx];
        if (res2) v += res2[idx];
        if (doRelu) v = fmaxf(v, 0.f);
        if (outF) outF[idx] = v;
        if (outB) outB[idx] = f2bf(v);
      }
}

// ---------- flash attention: 64 q-rows per block, 4 waves x 16 rows ----------

__global__ __launch_bounds__(128) void flash_attn_kernel(
    const __bf16* __restrict__ Q, const __bf16* __restrict__ Kb,
    const __bf16* __restrict__ V, const int* __restrict__ mask,
    __bf16* __restrict__ O)
{
  __shared__ __align__(16) __bf16 Qs[64 * 72];      // [q][dk]
  __shared__ __align__(16) __bf16 Ks[64 * 72];      // [key][dk]
  __shared__ __align__(16) __bf16 Vt[64 * 72];      // [dk][key]
  __shared__ __align__(16) __bf16 Ps[4][16 * 72];   // per-wave P [qlocal][key]

  const int tid = threadIdx.x, lane = tid & 31, wave = tid >> 5;
  const int qb = blockIdx.x * 64, h = blockIdx.y, b = blockIdx.z;
  const size_t qbase  = ((size_t)b * S_ + qb) * D_ + h * DK_;
  const size_t kvrow0 = (size_t)b * S_ * D_ + h * DK_;

  // per-thread 16B chunks of a 64x64 bf16 tile (512 chunks, 4 per thread)
  int rr[4], cc[4];
  unsigned qLds[4], kLds[4], tOff[4];
#pragma unroll
  for (int t = 0; t < 4; ++t) {
    const int c = tid + t * 128;
    rr[t] = c >> 3; cc[t] = c & 7;
    qLds[t] = lds_addr(&Qs[rr[t] * 72 + cc[t] * 8]);
    kLds[t] = lds_addr(&Ks[rr[t] * 72 + cc[t] * 8]);
    tOff[t] = (unsigned)((rr[t] * D_ + cc[t] * 8) * 2);
  }
  // Q tile: async copy, completion covered by first in-loop wait
#pragma unroll
  for (int t = 0; t < 4; ++t) async_copy16(qLds[t], tOff[t], Q + qbase);

  const int ho = (lane < 16) ? 0 : 8;
  const int nc = lane & 15;
  const int m0 = wave * 16;
  const v8f zero8 = {0.f, 0.f, 0.f, 0.f, 0.f, 0.f, 0.f, 0.f};

  v8f o[4];
  float m_run[8], l_run[8];
#pragma unroll
  for (int j = 0; j < 4; ++j) o[j] = zero8;
#pragma unroll
  for (int r = 0; r < 8; ++r) { m_run[r] = -3.0e38f; l_run[r] = 0.f; }

  for (int kt = 0; kt < 16; ++kt) {
    __syncthreads();  // previous iteration's reads of Ks/Vt complete
    const size_t kvb = kvrow0 + (size_t)kt * 64 * D_;
#pragma unroll
    for (int t = 0; t < 4; ++t) {
      async_copy16(kLds[t], tOff[t], Kb + kvb);           // K rows -> LDS direct
      v8bf vv = *(const v8bf*)(V + kvb + (size_t)rr[t] * D_ + cc[t] * 8);
#pragma unroll
      for (int e = 0; e < 8; ++e) Vt[(cc[t] * 8 + e) * 72 + rr[t]] = vv[e];
    }
    wait_async0();
    __syncthreads();

    // scores S = Q K^T : A = Qs rows, B = Ks rows (Bt layout)
    v8f s[4];
#pragma unroll
    for (int j = 0; j < 4; ++j) s[j] = zero8;
#pragma unroll
    for (int kk = 0; kk < 2; ++kk) {
      v16bf a = load_frag(Qs, m0, kk * 32, 72);
#pragma unroll
      for (int j = 0; j < 4; ++j) {
        v16bf bb = load_frag(Ks, j * 16, kk * 32, 72);
        s[j] = wmma_bf16(a, bb, s[j]);
      }
    }
    // scale 1/sqrt(64) + key mask
#pragma unroll
    for (int j = 0; j < 4; ++j) {
      const int km = mask[(size_t)b * S_ + kt * 64 + j * 16 + nc];
      const float madd = (km == 0) ? -1e30f : 0.f;
#pragma unroll
      for (int r = 0; r < 8; ++r) s[j][r] = s[j][r] * 0.125f + madd;
    }
    // online softmax (base-2); row stats per (half-wave, vgpr r)
    float alpha[8], rs[8];
#pragma unroll
    for (int r = 0; r < 8; ++r) {
      float mx = s[0][r];
#pragma unroll
      for (int j = 1; j < 4; ++j) mx = fmaxf(mx, s[j][r]);
#pragma unroll
      for (int off = 8; off >= 1; off >>= 1) mx = fmaxf(mx, __shfl_xor(mx, off, 32));
      const float mn = fmaxf(m_run[r], mx);
      alpha[r] = exp2f((m_run[r] - mn) * 1.44269504f);
      m_run[r] = mn;
      rs[r] = 0.f;
    }
#pragma unroll
    for (int j = 0; j < 4; ++j)
#pragma unroll
      for (int r = 0; r < 8; ++r) {
        const float p = exp2f((s[j][r] - m_run[r]) * 1.44269504f);
        s[j][r] = p; rs[r] += p;
      }
#pragma unroll
    for (int r = 0; r < 8; ++r) {
#pragma unroll
      for (int off = 8; off >= 1; off >>= 1) rs[r] += __shfl_xor(rs[r], off, 32);
      l_run[r] = l_run[r] * alpha[r] + rs[r];
    }
#pragma unroll
    for (int j = 0; j < 4; ++j)
#pragma unroll
      for (int r = 0; r < 8; ++r) o[j][r] *= alpha[r];

    // C->A relayout of P through per-wave LDS
    __bf16* P = &Ps[wave][0];
#pragma unroll
    for (int j = 0; j < 4; ++j)
#pragma unroll
      for (int r = 0; r < 8; ++r)
        P[(r + ho) * 72 + j * 16 + nc] = f2bf(s[j][r]);
    __syncthreads();

    // O += P @ V : A = P rows [16 x 64keys], B = Vt rows [dk][key]
#pragma unroll
    for (int kk = 0; kk < 2; ++kk) {
      v16bf a = load_frag(P, 0, kk * 32, 72);
#pragma unroll
      for (int j = 0; j < 4; ++j) {
        v16bf bb = load_frag(Vt, j * 16, kk * 32, 72);
        o[j] = wmma_bf16(a, bb, o[j]);
      }
    }
  }

#pragma unroll
  for (int j = 0; j < 4; ++j)
#pragma unroll
    for (int r = 0; r < 8; ++r) {
      const float val = o[j][r] / fmaxf(l_run[r], 1e-20f);
      const size_t row = (size_t)b * S_ + qb + m0 + r + ho;
      O[row * D_ + h * DK_ + j * 16 + nc] = f2bf(val);
    }
}

// ---------- launcher ----------

extern "C" void kernel_launch(void* const* d_in, const int* in_sizes, int n_in,
                              void* d_out, int out_size, void* d_ws, size_t ws_size,
                              hipStream_t stream)
{
  (void)in_sizes; (void)n_in; (void)out_size; (void)ws_size;
  const float* x    = (const float*)d_in[0];
  const float* kv   = (const float*)d_in[1];
  const int*   msk  = (const int*)  d_in[2];
  const float* ln1g = (const float*)d_in[3];
  const float* ln1b = (const float*)d_in[4];
  const float* wsrc[8] = { (const float*)d_in[5],  (const float*)d_in[6],
                           (const float*)d_in[7],  (const float*)d_in[8],
                           (const float*)d_in[10], (const float*)d_in[11],
                           (const float*)d_in[12], (const float*)d_in[13] };
  const float* ob0  = (const float*)d_in[9];
  const float* ob1  = (const float*)d_in[14];
  const float* ln2g = (const float*)d_in[15];
  const float* ln2b = (const float*)d_in[16];
  const float* w1   = (const float*)d_in[17];
  const float* b1   = (const float*)d_in[18];
  const float* w2   = (const float*)d_in[19];
  const float* b2   = (const float*)d_in[20];

  char* ws = (char*)d_ws;
  size_t off = 0;
  auto alloc = [&](size_t bytes) -> void* {
    void* p = ws + off;
    off += (bytes + 255) & ~(size_t)255;
    return p;
  };
  const size_t ROWS = (size_t)B_ * S_;  // 4096
  float*  y_f32  = (float*) alloc(ROWS * D_ * 4);
  __bf16* y_bf   = (__bf16*)alloc(ROWS * D_ * 2);
  __bf16* kvn_bf = (__bf16*)alloc(ROWS * D_ * 2);
  __bf16* wt[8];
  for (int i = 0; i < 8; ++i) wt[i] = (__bf16*)alloc((size_t)D_ * D_ * 2);
  __bf16* w1t    = (__bf16*)alloc((size_t)FFN_ * D_ * 2);
  __bf16* w2t    = (__bf16*)alloc((size_t)D_ * FFN_ * 2);
  __bf16* qB     = (__bf16*)alloc(ROWS * D_ * 2);
  __bf16* kB     = (__bf16*)alloc(ROWS * D_ * 2);
  __bf16* vB     = (__bf16*)alloc(ROWS * D_ * 2);
  __bf16* aB     = (__bf16*)alloc(ROWS * D_ * 2);
  float*  y1_f32 = (float*) alloc(ROWS * D_ * 4);
  __bf16* y1_bf  = (__bf16*)alloc(ROWS * D_ * 2);
  float*  x2_f32 = (float*) alloc(ROWS * D_ * 4);
  __bf16* x2n_bf = (__bf16*)alloc(ROWS * D_ * 2);
  __bf16* hB     = (__bf16*)alloc(ROWS * FFN_ * 2);

  // LayerNorm 1 on x (keep f32 for residual) and kv
  ln_kernel<<<(int)ROWS, 256, 0, stream>>>(x,  ln1g, ln1b, y_f32, y_bf);
  ln_kernel<<<(int)ROWS, 256, 0, stream>>>(kv, ln1g, ln1b, nullptr, kvn_bf);

  // weight conversion/transposition
  dim3 tb(32, 8);
  for (int i = 0; i < 8; ++i)
    transpose_f32_to_bf16<<<dim3(D_/32, D_/32), tb, 0, stream>>>(wsrc[i], wt[i], D_, D_);
  transpose_f32_to_bf16<<<dim3(FFN_/32, D_/32), tb, 0, stream>>>(w1, w1t, D_, FFN_);
  transpose_f32_to_bf16<<<dim3(D_/32, FFN_/32), tb, 0, stream>>>(w2, w2t, FFN_, D_);

  const dim3 gproj(D_/128, (unsigned)(ROWS/128));        // (8, 32)
  const dim3 gattn(S_/64, H_, B_);                       // (16,16,4)

  // cross-attention 0
  gemm_bf16_kernel<<<gproj, 256, 0, stream>>>(y_bf,   wt[0], nullptr, nullptr, nullptr, nullptr, qB, (int)ROWS, D_, D_, 0);
  gemm_bf16_kernel<<<gproj, 256, 0, stream>>>(kvn_bf, wt[1], nullptr, nullptr, nullptr, nullptr, kB, (int)ROWS, D_, D_, 0);
  gemm_bf16_kernel<<<gproj, 256, 0, stream>>>(kvn_bf, wt[2], nullptr, nullptr, nullptr, nullptr, vB, (int)ROWS, D_, D_, 0);
  flash_attn_kernel<<<gattn, 128, 0, stream>>>(qB, kB, vB, msk, aB);
  gemm_bf16_kernel<<<gproj, 256, 0, stream>>>(aB, wt[3], ob0, y_f32, nullptr, y1_f32, y1_bf, (int)ROWS, D_, D_, 0);

  // cross-attention 1 (query from y1)
  gemm_bf16_kernel<<<gproj, 256, 0, stream>>>(y1_bf,  wt[4], nullptr, nullptr, nullptr, nullptr, qB, (int)ROWS, D_, D_, 0);
  gemm_bf16_kernel<<<gproj, 256, 0, stream>>>(kvn_bf, wt[5], nullptr, nullptr, nullptr, nullptr, kB, (int)ROWS, D_, D_, 0);
  gemm_bf16_kernel<<<gproj, 256, 0, stream>>>(kvn_bf, wt[6], nullptr, nullptr, nullptr, nullptr, vB, (int)ROWS, D_, D_, 0);
  flash_attn_kernel<<<gattn, 128, 0, stream>>>(qB, kB, vB, msk, aB);
  // x2 = x + (o1 + ob1 + y1)
  gemm_bf16_kernel<<<gproj, 256, 0, stream>>>(aB, wt[7], ob1, y1_f32, x, x2_f32, nullptr, (int)ROWS, D_, D_, 0);

  // FFN
  ln_kernel<<<(int)ROWS, 256, 0, stream>>>(x2_f32, ln2g, ln2b, nullptr, x2n_bf);
  gemm_bf16_kernel<<<dim3(FFN_/128, (unsigned)(ROWS/128)), 256, 0, stream>>>(
      x2n_bf, w1t, b1, nullptr, nullptr, nullptr, hB, (int)ROWS, FFN_, D_, 1);
  gemm_bf16_kernel<<<gproj, 256, 0, stream>>>(
      hB, w2t, b2, x2_f32, nullptr, (float*)d_out, nullptr, (int)ROWS, D_, FFN_, 0);
}
